// DotAttn_80796924772504
// MI455X (gfx1250) — matrix operand order repaired and checked by reading, not compiled, not run on hardware
//
#include <hip/hip_runtime.h>
#include <hip/hip_bf16.h>

// ---------------------------------------------------------------------------
// Masked causal attention, fp32 in/out, bf16 WMMA compute (CDNA5 / gfx1250).
// Outputs (concatenated in d_out):  context (B,SQ,DV) then weights (B,SQ,SK).
//
// Kernel 1: weights = masked_softmax(QK^T/sqrt(D)); one wave per 16 q rows,
//           Q tile cached in LDS (bf16), K chunks double-buffered into LDS
//           via async global->LDS copies (no barriers: single-wave WG).
// Kernel 2: context = W @ V batched GEMM; 8-wave block, 64x64 tile, V tile
//           double-buffered into LDS via cooperative async copies.
// ---------------------------------------------------------------------------

#define B_   8
#define SQ_  2048
#define SK_  2048
#define D_   1024
#define DV_  1024

typedef __attribute__((ext_vector_type(16))) __bf16 v16bf;
typedef __attribute__((ext_vector_type(8)))  float  v8f;
typedef __attribute__((vector_size(16)))     int    i32x4;   // b128 payload

// ---- async global->LDS copy support (gfx1250 ASYNCcnt path) ----
#if defined(__has_builtin)
#if __has_builtin(__builtin_amdgcn_global_load_async_to_lds_b128) && \
    __has_builtin(__builtin_amdgcn_s_wait_asynccnt)
#define USE_ASYNC_LDS 1
#endif
#endif
#ifndef USE_ASYNC_LDS
#define USE_ASYNC_LDS 0
#endif

__device__ __forceinline__ void async_copy16(const float* g, float* l) {
#if USE_ASYNC_LDS
    __builtin_amdgcn_global_load_async_to_lds_b128(
        (__attribute__((address_space(1))) i32x4*)g,
        (__attribute__((address_space(3))) i32x4*)l,
        0, 0);
#else
    *(float4*)l = *(const float4*)g;
#endif
}

#if USE_ASYNC_LDS
#define WAIT_ASYNC(n) __builtin_amdgcn_s_wait_asynccnt(n)
#else
#define WAIT_ASYNC(n) ((void)0)
#endif

// ===========================================================================
// Kernel 1: weights = masked_softmax(Q K^T / sqrt(D))
// One wave (32 threads) per 16-row Q tile.  Grid = B * SQ/16 = 1024 blocks.
// ===========================================================================
__global__ __launch_bounds__(32)
void DotAttn_weights_kernel(const float* __restrict__ Q,
                            const float* __restrict__ K,
                            const int*   __restrict__ qlens,
                            const int*   __restrict__ klens,
                            float*       __restrict__ W)
{
    __shared__ __bf16 qs[16 * D_];          // 32 KB: Q tile cached as bf16
    __shared__ float  ks[2][16 * 32];       // 2 x 2 KB: K chunk double buffer

    const int lane = threadIdx.x & 31;
    const int grp  = lane >> 4;             // lane group 0/1 (WMMA layout)
    const int lrow = lane & 15;

    const int tile  = blockIdx.x;           // 0 .. B*SQ/16 - 1
    const int b     = tile / (SQ_ / 16);
    const int qbase = (tile % (SQ_ / 16)) * 16;

    const int qlen = qlens[b];
    const int klen = klens[b];

    // ---- stage Q tile to LDS (bf16) ----
    const float* Qb = Q + ((size_t)b * SQ_ + qbase) * (size_t)D_;
    for (int i = lane; i < 16 * D_; i += 32)
        qs[i] = (__bf16)Qb[i];
    __syncthreads();

    const float* Kb = K + (size_t)b * SK_ * (size_t)D_;
    float*       Wb = W + ((size_t)b * SQ_ + qbase) * (size_t)SK_;

    const float scale  = 0.03125f;          // 1/sqrt(1024)
    const float NEGBIG = -1.0e30f;          // masked marker

    // stage one 16(k-rows) x 32(d) fp32 K chunk into LDS buffer `buf`
    // lane copies 16 consecutive floats of one row: row = lane/2, col = 16*(lane&1)
    const int krow = lane >> 1;
    const int kcol16 = (lane & 1) * 16;
    auto stageK = [&](int buf, int chunk) {  // chunk = kt*32 + dc  flattened
        const int kt = chunk >> 5;
        const int d0 = (chunk & 31) * 32;
        const float* src = Kb + ((size_t)(kt * 16 + krow)) * D_ + d0 + kcol16;
        float* dst = &ks[buf][krow * 32 + kcol16];
        async_copy16(src,      dst);
        async_copy16(src + 4,  dst + 4);
        async_copy16(src + 8,  dst + 8);
        async_copy16(src + 12, dst + 12);
    };

    float mrow[8];
#pragma unroll
    for (int r = 0; r < 8; ++r) mrow[r] = NEGBIG;

    const int NCHUNK = (SK_ / 16) * (D_ / 32);  // 4096
    stageK(0, 0);

    // ---------------- Pass A: scores + mask + running max ----------------
    for (int kt = 0; kt < SK_ / 16; ++kt) {
        v8f acc = {0.f, 0.f, 0.f, 0.f, 0.f, 0.f, 0.f, 0.f};

        for (int dc = 0; dc < D_ / 32; ++dc) {
            const int c   = kt * (D_ / 32) + dc;
            const int cur = c & 1;
            if (c + 1 < NCHUNK) {
                stageK(cur ^ 1, c + 1);       // prefetch next chunk
                WAIT_ASYNC(4);                // retire current chunk's 4 copies
            } else {
                WAIT_ASYNC(0);
            }
            // single-wave WG: LDS ops from the same wave are in order; no barrier

            // A fragment (Q, 16x32 bf16) from LDS
            const int d0 = dc * 32;
            v16bf a;
            const __bf16* qa = qs + lrow * D_ + d0 + grp * 16;
#pragma unroll
            for (int i = 0; i < 8; ++i) { a[i] = qa[i]; a[i + 8] = qa[i + 8]; }

            // B fragment (K^T, 32x16 bf16): lane = column lrow of the tile,
            // 16 contiguous d values starting at grp*16 within the chunk.
            v16bf bb;
            const float* kp = &ks[cur][lrow * 32 + grp * 16];
#pragma unroll
            for (int i = 0; i < 16; ++i) bb[i] = (__bf16)kp[i];

            acc = __builtin_amdgcn_wmma_f32_16x16x32_bf16(
                      false, a, false, bb, (short)0, acc, false, false);
        }

        // mask, store raw scores, track row max
        const int kcol = kt * 16 + lrow;
#pragma unroll
        for (int r = 0; r < 8; ++r) {
            const int qrow   = qbase + r + grp * 8;
            const bool valid = (qrow < qlen) && (kcol < klen) && (kcol <= qrow);
            const float s    = valid ? acc[r] * scale : NEGBIG;

            float t = s;                                   // max over 16 lanes
            t = fmaxf(t, __shfl_xor(t, 1, 32));
            t = fmaxf(t, __shfl_xor(t, 2, 32));
            t = fmaxf(t, __shfl_xor(t, 4, 32));
            t = fmaxf(t, __shfl_xor(t, 8, 32));
            mrow[r] = fmaxf(mrow[r], t);

            Wb[(size_t)(r + grp * 8) * SK_ + kcol] = s;
        }
    }

    // ---------------- Pass B: exponentiate + row sums ----------------
    float lsum[8];
#pragma unroll
    for (int r = 0; r < 8; ++r) lsum[r] = 0.f;

    for (int kt = 0; kt < SK_ / 16; ++kt) {
        const int kcol = kt * 16 + lrow;
#pragma unroll
        for (int r = 0; r < 8; ++r) {
            const size_t idx = (size_t)(r + grp * 8) * SK_ + kcol;
            const float s = Wb[idx];
            const float e = (s <= -1.0e29f) ? 0.f : __expf(s - mrow[r]);
            float t = e;                                   // sum over 16 lanes
            t += __shfl_xor(t, 1, 32);
            t += __shfl_xor(t, 2, 32);
            t += __shfl_xor(t, 4, 32);
            t += __shfl_xor(t, 8, 32);
            lsum[r] += t;
            Wb[idx] = e;
        }
    }

    // ---------------- Pass C: normalize ----------------
    float inv[8];
#pragma unroll
    for (int r = 0; r < 8; ++r)
        inv[r] = (lsum[r] > 0.f) ? (1.f / lsum[r]) : 0.f;

    for (int kt = 0; kt < SK_ / 16; ++kt) {
        const int kcol = kt * 16 + lrow;
#pragma unroll
        for (int r = 0; r < 8; ++r) {
            const size_t idx = (size_t)(r + grp * 8) * SK_ + kcol;
            Wb[idx] = Wb[idx] * inv[r];
        }
    }
}

// ===========================================================================
// Kernel 2: context = W x V   (per batch: 2048x2048 @ 2048x1024, bf16 WMMA)
// Block = 256 threads (8 waves), block tile 64(M) x 64(N),
// wave tile 16(M) x 32(N).  V tile (32x64 fp32) double-buffered in LDS via
// cooperative async copies.  Grid = (DV/64, SQ/64, B) = (16, 32, 8).
// ===========================================================================
__global__ __launch_bounds__(256)
void DotAttn_context_kernel(const float* __restrict__ W,
                            const float* __restrict__ V,
                            float*       __restrict__ O)
{
    __shared__ float vs[2][32][64];         // 16 KB double-buffered V tile

    const int tid  = threadIdx.x;
    const int lane = tid & 31;
    const int wid  = tid >> 5;              // 0..7
    const int grp  = lane >> 4;
    const int lrow = lane & 15;

    const int b    = blockIdx.z;
    const int m0   = blockIdx.y * 64 + (wid & 3) * 16;
    const int nblk = blockIdx.x * 64;
    const int n0   = (wid >> 2) * 32;       // N offset within the block tile

    const float* Wb = W + (size_t)b * SQ_ * SK_;
    const float* Vb = V + (size_t)b * SK_ * DV_ + nblk;

    // cooperative copy slots: 2 x b128 per thread per 32x64 chunk
    const int e0 = tid * 4;                 // element 0..1020
    const int r0 = e0 >> 6, c0 = e0 & 63;
    const int e1 = e0 + 1024;
    const int r1 = e1 >> 6, c1 = e1 & 63;

    auto stageV = [&](int buf, int k0) {
        async_copy16(Vb + (size_t)(k0 + r0) * DV_ + c0, &vs[buf][r0][c0]);
        async_copy16(Vb + (size_t)(k0 + r1) * DV_ + c1, &vs[buf][r1][c1]);
    };

    v8f acc0 = {0.f, 0.f, 0.f, 0.f, 0.f, 0.f, 0.f, 0.f};
    v8f acc1 = {0.f, 0.f, 0.f, 0.f, 0.f, 0.f, 0.f, 0.f};

    const float* wrow = Wb + (size_t)(m0 + lrow) * SK_;

    const int NCH = SK_ / 32;               // 64 k-chunks
    stageV(0, 0);

    for (int kc = 0; kc < NCH; ++kc) {
        const int cur = kc & 1;
        if (kc + 1 < NCH) {
            stageV(cur ^ 1, (kc + 1) * 32); // prefetch next chunk
            WAIT_ASYNC(2);                  // retire current chunk's copies
        } else {
            WAIT_ASYNC(0);
        }
        __syncthreads();                    // all waves' slices visible

        // A fragment (weights row, 16x32 bf16) from global (L2-resident)
        const float* ap = wrow + kc * 32 + grp * 16;
        if (kc + 1 < NCH)
            __builtin_prefetch(wrow + (kc + 1) * 32, 0, 0);
        v16bf a;
#pragma unroll
        for (int i = 0; i < 8; ++i) {
            a[i]     = (__bf16)ap[i];
            a[i + 8] = (__bf16)ap[i + 8];
        }

        // B fragments (V, 32x16 bf16) from LDS
        v16bf b0v, b1v;
#pragma unroll
        for (int i = 0; i < 16; ++i) {
            b0v[i] = (__bf16)vs[cur][grp * 16 + i][n0 + lrow];
            b1v[i] = (__bf16)vs[cur][grp * 16 + i][n0 + lrow + 16];
        }

        acc0 = __builtin_amdgcn_wmma_f32_16x16x32_bf16(
                   false, a, false, b0v, (short)0, acc0, false, false);
        acc1 = __builtin_amdgcn_wmma_f32_16x16x32_bf16(
                   false, a, false, b1v, (short)0, acc1, false, false);

        __syncthreads();                    // done reading before buffer reuse
    }

    float* orow = O + ((size_t)b * SQ_ + m0) * (size_t)DV_;
#pragma unroll
    for (int r = 0; r < 8; ++r) {
        orow[(size_t)(r + grp * 8) * DV_ + nblk + n0 + lrow]      = acc0[r];
        orow[(size_t)(r + grp * 8) * DV_ + nblk + n0 + 16 + lrow] = acc1[r];
    }
}

// ===========================================================================
extern "C" void kernel_launch(void* const* d_in, const int* in_sizes, int n_in,
                              void* d_out, int out_size, void* d_ws, size_t ws_size,
                              hipStream_t stream) {
    const float* Q  = (const float*)d_in[0];
    const float* K  = (const float*)d_in[1];
    const float* V  = (const float*)d_in[2];
    const int*   ql = (const int*)d_in[3];
    const int*   kl = (const int*)d_in[4];

    float* ctx = (float*)d_out;                              // (B,SQ,DV)
    float* wts = ctx + (size_t)B_ * SQ_ * DV_;               // (B,SQ,SK)

    // 1) weights = masked_softmax(QK^T/sqrt(D))
    DotAttn_weights_kernel<<<B_ * (SQ_ / 16), 32, 0, stream>>>(Q, K, ql, kl, wts);

    // 2) context = weights @ V
    dim3 grid2(DV_ / 64, SQ_ / 64, B_);
    DotAttn_context_kernel<<<grid2, 256, 0, stream>>>(wts, V, ctx);
}
